// ConvLSTM_43688407335591
// MI455X (gfx1250) — matrix operand order repaired and checked by reading, not compiled, not run on hardware
//
#include <hip/hip_runtime.h>

// ---------------------------------------------------------------------------
// GRU-style recurrence on MI455X (gfx1250, wave32, WMMA, TDM):
//   Phase 1 (parallel):  CCX = bf16(x) @ Wfc[:, :D]^T   [32768 x 2048]
//                        C2X = bf16(x) @ Wfc2[:, :D]^T  [32768 x 1024]
//   Phase 2: ONE persistent kernel, 16 resident workgroups, 512 iterations:
//        TDM: stage h (bf16, 128KB) into LDS (padded pitch, bank-friendly)
//        cc = h @ Wfc[:, D:]^T + CCX[t]  -> z, r*h      (device barrier)
//        TDM: stage r*h into LDS
//        s  = (r*h) @ Wfc2[:, D:]^T + C2X[t]
//        h  = (1-z)*h + z*tanh(s)                       (device barrier)
//   A-operand fragments come from LDS (ds_load_b128) instead of being
//   replayed 8x per block through L0; weights stream from L2 (6 MB, hot).
// ---------------------------------------------------------------------------

typedef __bf16 bf16;
typedef __attribute__((ext_vector_type(16))) __bf16        v16bf;
typedef __attribute__((ext_vector_type(8)))  float         v8f;
typedef __attribute__((ext_vector_type(4)))  unsigned int  u32x4;
typedef __attribute__((ext_vector_type(8)))  int           i32x8;
typedef __attribute__((ext_vector_type(4)))  int           i32x4;

union Frag16 { v16bf v; u32x4 q[2]; };

// A fragment: 16x32 bf16, row-major source. Lane L<16 -> row M=L, K {0..7,16..23};
// lane L>=16 -> row M=L-16, K {8..15,24..31}  (ISA 7.12.2, 16-bit A 16x32).
__device__ __forceinline__ v16bf load_A(const bf16* __restrict__ p) {
  Frag16 f;
  f.q[0] = *(const u32x4*)(p);        // K = kbase .. kbase+7
  f.q[1] = *(const u32x4*)(p + 16);   // K = kbase+16 .. kbase+23
  return f.v;
}

// B fragment: 32(K)x16(N) bf16 from W[N,K] row-major (B[k,n] = W[n,k]).
__device__ __forceinline__ v16bf load_B(const bf16* __restrict__ p) {
  Frag16 f;
  f.q[0] = *(const u32x4*)(p);
  f.q[1] = *(const u32x4*)(p + 16);
  return f.v;
}

__device__ __forceinline__ v8f wmma_bf16(v16bf a, v16bf b, v8f c) {
  return __builtin_amdgcn_wmma_f32_16x16x32_bf16(false, a, false, b,
                                                 (short)0, c, false, false);
}

__device__ __forceinline__ float sigmoidf_(float x) {
  return 1.0f / (1.0f + __expf(-x));
}

// ---- LDS staging geometry (TDM pad: +4 dwords every 256 dwords) ----
// Row = 1024 bf16 = 2048 B = two 1024B intervals -> pitch = 2048 + 2*16 = 2080 B.
// Row pitch of 520 dwords shifts consecutive rows by 8 banks (2-way worst case
// for b128 reads across 16 rows, vs 16-way with an unpadded 2048B pitch).
#define LDS_PITCH   2080
#define LDS_BYTES   (64 * LDS_PITCH)   // 133,120 B (< 320KB/WGP)

__device__ __forceinline__ const bf16* lds_elem(const bf16* base, int m, int k) {
  return (const bf16*)((const char*)base + m * LDS_PITCH + k * 2 +
                       ((k >= 512) ? 16 : 0));
}

// Issue one TDM 2-D tile load: 64 rows x 1024 bf16 (row stride 1024) -> LDS 0,
// with pad_enable so the LDS image has the 2080B pitch above.
// D# layout per CDNA5 ISA.
__device__ __forceinline__ void tdm_stage_h(const bf16* gsrc) {
  unsigned long long ga = (unsigned long long)(size_t)gsrc;
  u32x4 g0;
  g0[0] = 1u;                                   // count=1 (valid user D#)
  g0[1] = 0u;                                   // lds_addr = 0 (dynamic LDS base)
  g0[2] = (unsigned)(ga & 0xFFFFFFFFu);         // global_addr[31:0]
  g0[3] = (unsigned)((ga >> 32) & 0x1FFFFFFu)   // global_addr[56:32]
        | (2u << 30);                           // type=2 ("image")
  i32x8 g1;
  g1[0] = (1 << 16)                             // data_size = 1 -> 2 bytes
        | (1 << 20)                             // pad_enable
        | (7 << 22)                             // pad_interval: 256 dwords
        | (3 << 25);                            // pad_amount:   4 dwords
  g1[1] = (1024 & 0xFFFF) << 16;                // tensor_dim0 lo16 (=1024)
  g1[2] = (1024 >> 16) | (64 << 16);            // tensor_dim0 hi16 | tensor_dim1 lo16
  g1[3] = (1024 << 16);                         // tensor_dim1 hi16 | tile_dim0=1024
  g1[4] = 64;                                   // tile_dim1=64, tile_dim2=0
  g1[5] = 1024;                                 // tensor_dim0_stride lo32
  g1[6] = 0;                                    // stride0 hi16 | stride1 lo16
  g1[7] = 0;
  i32x4 z4 = {0, 0, 0, 0};                      // groups 2/3: 2-D tensor, unused
#if __clang_major__ >= 23
  i32x8 z8 = {0, 0, 0, 0, 0, 0, 0, 0};
  __builtin_amdgcn_tensor_load_to_lds(g0, g1, z4, z4, z8, 0);
#else
  __builtin_amdgcn_tensor_load_to_lds(g0, g1, z4, z4, 0);
#endif
}

// One wave computes a (MT*16)(M) x 16(N) strip, A from global memory.
template <int MT>
__device__ __forceinline__ void mmNx16(const bf16* __restrict__ A,
                                       const bf16* __restrict__ W,
                                       int m0, int n0, int K, v8f acc[MT]) {
  const int lane = threadIdx.x & 31;
  const int hi   = lane >> 4;
  const bf16* ap = A + (size_t)(m0 + (lane & 15)) * K + (hi << 3);
  const bf16* bp = W + (size_t)(n0 + (lane & 15)) * K + (hi << 3);
  const size_t a16 = (size_t)16 * K;
  for (int k = 0; k < K; k += 32) {
    v16bf b = load_B(bp + k);
#pragma unroll
    for (int i = 0; i < MT; ++i) {
      v16bf a = load_A(ap + k + (size_t)i * a16);
      acc[i] = wmma_bf16(a, b, acc[i]);
    }
  }
}

// Same, but A fragments come from the padded LDS image (ds_load_b128).
template <int MT>
__device__ __forceinline__ void mmNx16_ldsA(const bf16* Alds,
                                            const bf16* __restrict__ W,
                                            int m0, int n0, int K, v8f acc[MT]) {
  const int lane = threadIdx.x & 31;
  const int hi   = lane >> 4;
  const int ar   = m0 + (lane & 15);
  const int ka   = hi << 3;             // A chunk K-offset {0,8}
  const bf16* bp = W + (size_t)(n0 + (lane & 15)) * K + (hi << 3);
  for (int k = 0; k < K; k += 32) {
    v16bf b = load_B(bp + k);
#pragma unroll
    for (int i = 0; i < MT; ++i) {
      Frag16 f;
      f.q[0] = *(const u32x4*)lds_elem(Alds, ar + i * 16, k + ka);
      f.q[1] = *(const u32x4*)lds_elem(Alds, ar + i * 16, k + ka + 16);
      acc[i] = wmma_bf16(f.v, b, acc[i]);
    }
  }
}

// Device-scope sense-reversing barrier across NB resident workgroups.
template <int NB>
__device__ __forceinline__ void grid_sync(unsigned* bar) {
  __threadfence();
  __syncthreads();
  if (threadIdx.x == 0) {
    unsigned g = __hip_atomic_load(bar + 1, __ATOMIC_RELAXED,
                                   __HIP_MEMORY_SCOPE_AGENT);
    unsigned a = __hip_atomic_fetch_add(bar, 1u, __ATOMIC_ACQ_REL,
                                        __HIP_MEMORY_SCOPE_AGENT);
    if (a == (unsigned)(NB - 1)) {
      __hip_atomic_store(bar, 0u, __ATOMIC_RELAXED, __HIP_MEMORY_SCOPE_AGENT);
      __hip_atomic_store(bar + 1, g + 1u, __ATOMIC_RELEASE,
                         __HIP_MEMORY_SCOPE_AGENT);
    } else {
      while (__hip_atomic_load(bar + 1, __ATOMIC_ACQUIRE,
                               __HIP_MEMORY_SCOPE_AGENT) == g)
        __builtin_amdgcn_s_sleep(1);
    }
  }
  __syncthreads();
  __threadfence();
}

// ---------------- conversion / init kernels ----------------

__global__ void __launch_bounds__(256) k_cvt(const float* __restrict__ in,
                                             bf16* __restrict__ out, int n) {
  for (int i = blockIdx.x * blockDim.x + threadIdx.x; i < n;
       i += gridDim.x * blockDim.x)
    out[i] = (bf16)in[i];
}

__global__ void __launch_bounds__(256) k_split(const float* __restrict__ W,
                                               bf16* __restrict__ Wx,
                                               bf16* __restrict__ Wh, int total) {
  for (int i = blockIdx.x * blockDim.x + threadIdx.x; i < total;
       i += gridDim.x * blockDim.x) {
    int r = i >> 11, c = i & 2047;
    bf16 v = (bf16)W[i];
    if (c < 1024) Wx[(r << 10) + c] = v;
    else          Wh[(r << 10) + (c - 1024)] = v;
  }
}

__global__ void __launch_bounds__(256) k_init(float* __restrict__ hbuf,
                                              bf16* __restrict__ hbf,
                                              unsigned* __restrict__ bar) {
  int i = blockIdx.x * blockDim.x + threadIdx.x;
  if (i < 64 * 1024) { hbuf[i] = 1e-9f; hbf[i] = (bf16)1e-9f; }
  if (i < 2) bar[i] = 0u;   // re-arm device barrier every call (graph-replay safe)
}

// ---------------- Phase 1: big parallel x-projection GEMM ----------------
__global__ void __launch_bounds__(256) k_gemm_xproj(const bf16* __restrict__ A,
                                                    const bf16* __restrict__ W,
                                                    bf16* __restrict__ C,
                                                    int N, int K) {
  const int wave = threadIdx.x >> 5;
  const int lane = threadIdx.x & 31;
  const int n0 = blockIdx.x * 128 + wave * 16;
  const int m0 = blockIdx.y * 64;
  v8f acc[4] = {};
  mmNx16<4>(A, W, m0, n0, K, acc);
  const int hi8 = (lane >> 4) << 3;
  const int col = n0 + (lane & 15);
  for (int i = 0; i < 4; ++i)
    for (int j = 0; j < 8; ++j)
      C[(size_t)(m0 + i * 16 + hi8 + j) * N + col] = (bf16)acc[i][j];
}

// ---------------- Phase 2: persistent recurrent kernel ----------------
// 16 workgroups x 256 threads, 130KB dynamic LDS each; all co-resident.
__global__ void __launch_bounds__(256) k_recurrent(
    const bf16* __restrict__ Wh1,   // [2048,1024]
    const bf16* __restrict__ Wh2,   // [1024,1024]
    const bf16* __restrict__ CCX,   // [512*64, 2048]
    const bf16* __restrict__ C2X,   // [512*64, 1024]
    float* __restrict__ hbuf,       // [64,1024]
    bf16*  __restrict__ hbf,        // [64,1024]
    float* __restrict__ zbuf,       // [64,1024]
    bf16*  __restrict__ rhbf,       // [64,1024]
    float* __restrict__ out,        // [512,64,1024]
    unsigned* __restrict__ bar) {
  extern __shared__ char smem[];
  const bf16* lds_a = (const bf16*)smem;
  const int wave = threadIdx.x >> 5;
  const int lane = threadIdx.x & 31;
  const int hi8  = (lane >> 4) << 3;
  const int lcol = lane & 15;

  for (int t = 0; t < 512; ++t) {
    const bf16* ccx_t = CCX + (size_t)t * 64 * 2048;
    const bf16* c2x_t = C2X + (size_t)t * 64 * 1024;
    // Pull next step's x-projection slices toward the caches during the GEMMs.
    __builtin_prefetch(ccx_t + 64 * 2048 +
                       (size_t)(blockIdx.x * 256 + threadIdx.x) * 32, 0, 1);
    __builtin_prefetch(c2x_t + 64 * 1024 +
                       (size_t)(blockIdx.x * 256 + threadIdx.x) * 16, 0, 1);

    // ---- TDM: stage h tile into LDS, then gates ----
    if (wave == 0) {
      tdm_stage_h(hbf);
      __builtin_amdgcn_s_wait_tensorcnt(0);
    }
    __syncthreads();
    {
      const int n0 = blockIdx.x * 128 + wave * 16;   // 128 waves x 16 = 2048 cols
      v8f acc[4] = {};
      mmNx16_ldsA<4>(lds_a, Wh1, 0, n0, 1024, acc);
      const int col = n0 + lcol;
#pragma unroll
      for (int i = 0; i < 4; ++i)
#pragma unroll
        for (int j = 0; j < 8; ++j) {
          int m = i * 16 + hi8 + j;
          float v = acc[i][j] + (float)ccx_t[(size_t)m * 2048 + col];
          if (col < 1024) {                          // z gate
            zbuf[m * 1024 + col] = sigmoidf_(v);
          } else {                                   // r gate -> r*h (bf16)
            int c2 = col - 1024;
            float r = sigmoidf_(v);
            rhbf[m * 1024 + c2] = (bf16)(r * hbuf[m * 1024 + c2]);
          }
        }
    }
    grid_sync<16>(bar);

    // ---- TDM: stage r*h tile into LDS, then candidate + state update ----
    if (wave == 0) {
      tdm_stage_h(rhbf);
      __builtin_amdgcn_s_wait_tensorcnt(0);
    }
    __syncthreads();
    {
      const int gw = blockIdx.x * 8 + wave;          // 0..127
      const int n0 = (gw >> 1) * 16;                 // 64 col strips
      const int m0 = (gw & 1) * 32;                  // M split in half
      v8f acc[2] = {};
      mmNx16_ldsA<2>(lds_a, Wh2, m0, n0, 1024, acc);
      float* out_t = out + (size_t)t * 64 * 1024;
      const int col = n0 + lcol;
#pragma unroll
      for (int i = 0; i < 2; ++i)
#pragma unroll
        for (int j = 0; j < 8; ++j) {
          int m = m0 + i * 16 + hi8 + j;
          int idx = m * 1024 + col;
          float s  = tanhf(acc[i][j] + (float)c2x_t[idx]);
          float z  = zbuf[idx];
          float hn = (1.0f - z) * hbuf[idx] + z * s;
          hbuf[idx]  = hn;
          hbf[idx]   = (bf16)hn;
          out_t[idx] = hn;
        }
    }
    grid_sync<16>(bar);
  }
}

// ---------------------------------------------------------------------------

extern "C" void kernel_launch(void* const* d_in, const int* in_sizes, int n_in,
                              void* d_out, int out_size, void* d_ws, size_t ws_size,
                              hipStream_t stream) {
  constexpr int T = 512, B = 64, D = 1024, H = 1024;
  constexpr int M = T * B;          // 32768 rows of x
  const float* x    = (const float*)d_in[0];   // [T,B,D]
  const float* Wfc  = (const float*)d_in[1];   // [2H, D+H]
  const float* Wfc2 = (const float*)d_in[2];   // [H,  D+H]
  float* out = (float*)d_out;                  // [T,B,H]

  char* w = (char*)d_ws;
  auto carve = [&](size_t bytes) {
    char* p = w;
    w += (bytes + 255) & ~(size_t)255;
    return p;
  };
  bf16*  Xbf  = (bf16*)carve((size_t)M * D * 2);        //  64 MB
  bf16*  Wx1  = (bf16*)carve((size_t)2 * H * D * 2);    //   4 MB
  bf16*  Wh1  = (bf16*)carve((size_t)2 * H * H * 2);    //   4 MB
  bf16*  Wx2  = (bf16*)carve((size_t)H * D * 2);        //   2 MB
  bf16*  Wh2  = (bf16*)carve((size_t)H * H * 2);        //   2 MB
  bf16*  CCX  = (bf16*)carve((size_t)M * 2 * H * 2);    // 128 MB
  bf16*  C2X  = (bf16*)carve((size_t)M * H * 2);        //  64 MB (follows CCX: prefetch stays in-bounds)
  float* hbuf = (float*)carve((size_t)B * H * 4);
  bf16*  hbf  = (bf16*)carve((size_t)B * H * 2);
  float* zbuf = (float*)carve((size_t)B * H * 4);
  bf16*  rhbf = (bf16*)carve((size_t)B * H * 2);
  unsigned* bar = (unsigned*)carve(256);
  (void)ws_size; (void)n_in; (void)in_sizes; (void)out_size;

  // ---- conversions (f32 -> bf16), weight split into x/h halves ----
  k_cvt  <<<2048, 256, 0, stream>>>(x, Xbf, M * D);
  k_split<<<1024, 256, 0, stream>>>(Wfc,  Wx1, Wh1, 2 * H * (D + H));
  k_split<<< 512, 256, 0, stream>>>(Wfc2, Wx2, Wh2, H * (D + H));
  k_init <<<256, 256, 0, stream>>>(hbuf, hbf, bar);

  // ---- Phase 1: parallel x-projections (all 512 timesteps at once) ----
  k_gemm_xproj<<<dim3((2 * H) / 128, M / 64), 256, 0, stream>>>(Xbf, Wx1, CCX, 2 * H, D);
  k_gemm_xproj<<<dim3(H / 128,       M / 64), 256, 0, stream>>>(Xbf, Wx2, C2X, H, D);

  // ---- Phase 2: one persistent kernel runs the whole 512-step recurrence ----
  k_recurrent<<<dim3(16), 256, LDS_BYTES, stream>>>(Wh1, Wh2, CCX, C2X,
                                                    hbuf, hbf, zbuf, rhbf,
                                                    out, bar);
}
